// MultilayerGRU_71725953843286
// MI455X (gfx1250) — matrix operand order repaired and checked by reading, not compiled, not run on hardware
//
#include <hip/hip_runtime.h>
#include <hip/hip_bf16.h>

// ---------------------------------------------------------------------------
// Multilayer GRU (B=32, S=512, I=H=O=1024, L=2) for gfx1250 (MI455X).
// bf16 WMMA (v_wmma_f32_16x16x32_bf16), fp32 accumulate.
// - Batched GEMMs (XP/Y): A tile staged to LDS via global_load_async_to_lds,
//   XOR-swizzled, 16x64 wave tiles, fully unrolled ping-pong K loop.
// - Serial recurrence (ZR/G): direct L2-resident loads, 16x16 wave tiles,
//   maximum wave parallelism for the latency-critical path.
// ---------------------------------------------------------------------------

typedef __attribute__((ext_vector_type(16))) __bf16 v16bf;
typedef __attribute__((ext_vector_type(8)))  __bf16 v8bf;
typedef __attribute__((ext_vector_type(8)))  float  v8f;

#define DEV __device__ __forceinline__

constexpr int B_ = 32, S_ = 512, I_ = 1024, H_ = 1024, L_ = 2, O_ = 1024;
constexpr int H3 = 3 * H_;
constexpr int KC = 1024;            // every GEMM here has K = 1024
constexpr int AROW = 1024;          // A row length (elements) for all A operands

// ---- WMMA fragment helpers (ISA 7.12.2 layouts) ---------------------------
// 16-bit A 16x32: lanes 0-15 hold K = {0..7, 16..23}, lanes 16-31 hold
// K = {8..15, 24..31}: per lane two contiguous 16B chunks, 32B apart.
DEV v16bf combine_a(v8bf lo, v8bf hi) {
  v16bf a;
#pragma unroll
  for (int i = 0; i < 8; ++i) { a[i] = lo[i]; a[i + 8] = hi[i]; }
  return a;
}

// LDS A tile: 32 rows x 2048B, XOR-swizzled at 16B granularity so that
// row-major fragment reads (row stride 2048B == bank-period multiple) spread
// across all 64 banks: phys_granule = granule ^ (row & 15).
DEV v16bf load_a_lds(const __bf16* Alds, int row, int k0, int half) {
  const int rs = row & 15;
  const char* rowp = (const char*)Alds + (row << 11);
  const int g = (k0 >> 3) + half;                      // 16B granule index
  v8bf lo = *reinterpret_cast<const v8bf*>(rowp + (((g    ) ^ rs) << 4));
  v8bf hi = *reinterpret_cast<const v8bf*>(rowp + (((g + 2) ^ rs) << 4));
  return combine_a(lo, hi);
}

// Stage one 32x1024 bf16 A-tile (64KB) global -> LDS with async copies.
// 256 threads x 16 transfers x 16B = 64KB, tracked on ASYNCcnt.
DEV void stage_a_async(const __bf16* __restrict__ Ag, __bf16* Alds) {
  const int tid = threadIdx.x;
  const unsigned lbase = (unsigned)(size_t)Alds;       // LDS byte offset
#pragma unroll
  for (int i = 0; i < 16; ++i) {
    const int idx = tid + i * 256;                     // 0..4095 granules
    const int row = idx >> 7;                          // 128 granules/row
    const int cc  = idx & 127;
    const char* gp = (const char*)Ag + ((size_t)row << 11) + (cc << 4);
    const unsigned loff =
        lbase + (unsigned)((row << 11) + ((cc ^ (row & 15)) << 4));
    asm volatile("global_load_async_to_lds_b128 %0, %1, off"
                 :: "v"(loff), "v"(gp) : "memory");
  }
  asm volatile("s_wait_asynccnt 0" ::: "memory");
  __syncthreads();
}

// Multi-subtile WMMA core: one wave computes 16 x (16*NW) of
// D = A[16 rows, K] * W[n0:n0+16*NW, :K]^T. K compile-time, fully unrolled
// ping-pong (no cross-iteration register copies). B streamed from global
// (weights, L2-resident); A from LDS (LDSA) or global.
template <int NW, bool LDSA>
DEV void wmma_core(const __bf16* __restrict__ A,   // LDS base or global base
                   int arow_idx,                   // row in tile (LDSA) / abs row
                   const __bf16* __restrict__ W, int ldw,
                   int n0, int lane, v8f acc[NW]) {
  const int half = lane >> 4;
  const int l15  = lane & 15;

  auto loadA = [&](int k0) -> v16bf {
    if constexpr (LDSA) {
      return load_a_lds(A, arow_idx, k0, half);
    } else {
      const __bf16* p = A + (size_t)arow_idx * AROW + k0 + half * 8;
      return combine_a(*reinterpret_cast<const v8bf*>(p),
                       *reinterpret_cast<const v8bf*>(p + 16));
    }
  };

  const __bf16* brow[NW];
#pragma unroll
  for (int j = 0; j < NW; ++j)
    brow[j] = W + (size_t)(n0 + j * 16 + l15) * ldw + half * 8;

  v16bf b0[NW], b1[NW];
#pragma unroll
  for (int j = 0; j < NW; ++j)
    b0[j] = combine_a(*reinterpret_cast<const v8bf*>(brow[j]),
                      *reinterpret_cast<const v8bf*>(brow[j] + 16));

#pragma unroll
  for (int k0 = 0; k0 < KC; k0 += 64) {
    // load the k0+32 B fragments while computing on k0
#pragma unroll
    for (int j = 0; j < NW; ++j) {
      b1[j] = combine_a(*reinterpret_cast<const v8bf*>(brow[j] + k0 + 32),
                        *reinterpret_cast<const v8bf*>(brow[j] + k0 + 48));
      if (k0 + 160 < KC)  // stream weights ahead -> global_prefetch
        __builtin_prefetch((const void*)(brow[j] + k0 + 160), 0, 3);
    }
    {
      v16bf a = loadA(k0);
#pragma unroll
      for (int j = 0; j < NW; ++j)
        acc[j] = __builtin_amdgcn_wmma_f32_16x16x32_bf16(
            false, a, false, b0[j], (short)0, acc[j], false, false);
    }
    // load the k0+64 B fragments while computing on k0+32
    if (k0 + 64 < KC) {
#pragma unroll
      for (int j = 0; j < NW; ++j)
        b0[j] = combine_a(*reinterpret_cast<const v8bf*>(brow[j] + k0 + 64),
                          *reinterpret_cast<const v8bf*>(brow[j] + k0 + 80));
    }
    {
      v16bf a = loadA(k0 + 32);
#pragma unroll
      for (int j = 0; j < NW; ++j)
        acc[j] = __builtin_amdgcn_wmma_f32_16x16x32_bf16(
            false, a, false, b1[j], (short)0, acc[j], false, false);
    }
  }
}

DEV float sigmoidf_(float x) { return 1.f / (1.f + __expf(-x)); }

// ---- fused GEMM kernel with mode-specific epilogues -----------------------
enum GemmMode { MODE_XP = 0, MODE_ZR = 1, MODE_G = 2, MODE_Y = 3 };

// Block = 256 threads = 8 waves (2 M-rows x 4 N-groups).
// Block tile: 32(M) x (64*NW)(N); wave tile 16 x (16*NW).
template <int MODE, int NW>
__global__ __launch_bounds__(256) void gru_wmma_gemm(
    const __bf16* __restrict__ A,     // [M,1024] bf16 activations
    const __bf16* __restrict__ W,     // [N,1024] bf16 weights (row-major)
    const float* __restrict__ bias,   // bh + l*3H (ZR/G) or by (Y)
    const float* __restrict__ xp,     // XP [B*S, 3H] fp32 (ZR/G)
    float*  __restrict__ xp_out,      // XP dest (MODE_XP)
    float*  __restrict__ zbuf,        // z gate [B,H] fp32
    float*  __restrict__ hf,          // h state [B,H] fp32
    __bf16* __restrict__ hb,          // h state [B,H] bf16 (next-step A)
    __bf16* __restrict__ rhb,         // r*h     [B,H] bf16 (G-step A)
    __bf16* __restrict__ seq,         // sequence buffer [B*S, H] bf16
    float*  __restrict__ yout,        // d_out (MODE_Y)
    int t) {
  constexpr bool LDSA = (MODE == MODE_XP || MODE == MODE_Y);
  __shared__ __bf16 Alds[LDSA ? 32 * AROW : 16];  // 64KB when staged

  const int lane = threadIdx.x & 31;
  const int wave = threadIdx.x >> 5;
  const int mtile = (wave & 1) * 16;               // M sub-tile within block
  const int m0 = blockIdx.y * 32 + mtile;
  const int n0 = blockIdx.x * (64 * NW) + (wave >> 1) * (16 * NW);

  if constexpr (LDSA)
    stage_a_async(A + ((size_t)blockIdx.y * 32) * AROW, Alds);

  v8f acc[NW];
#pragma unroll
  for (int j = 0; j < NW; ++j)
    acc[j] = v8f{0.f, 0.f, 0.f, 0.f, 0.f, 0.f, 0.f, 0.f};

  const int arow_idx = LDSA ? (mtile + (lane & 15)) : (m0 + (lane & 15));
  wmma_core<NW, LDSA>(LDSA ? Alds : A, arow_idx, W, KC, n0, lane, acc);

  const int half = lane >> 4;
#pragma unroll
  for (int j = 0; j < NW; ++j) {
    const int col = n0 + j * 16 + (lane & 15);
#pragma unroll
    for (int e = 0; e < 8; ++e) {
      const int row = m0 + half * 8 + e;  // C/D: vgpr e -> M = e + 8*(lane>=16)
      float v = acc[j][e];
      if constexpr (MODE == MODE_XP) {
        xp_out[(size_t)row * H3 + col] = v;
      } else if constexpr (MODE == MODE_ZR) {
        // row = batch b; col in [0, 2H): z gate then r gate
        const float val = v + bias[col] + xp[((size_t)row * S_ + t) * H3 + col];
        const float s = sigmoidf_(val);
        if (col < H_) {
          zbuf[row * H_ + col] = s;
        } else {
          const int nh = col - H_;
          rhb[row * H_ + nh] = (__bf16)(s * hf[row * H_ + nh]);
        }
      } else if constexpr (MODE == MODE_G) {
        // row = batch b; col in [0, H)
        const float val = v + bias[2 * H_ + col] +
                          xp[((size_t)row * S_ + t) * H3 + 2 * H_ + col];
        const float g = tanhf(val);
        const float z = zbuf[row * H_ + col];
        const float hprev = hf[row * H_ + col];
        const float hnew = z * hprev + (1.f - z) * g;
        hf[row * H_ + col] = hnew;
        hb[row * H_ + col] = (__bf16)hnew;
        seq[((size_t)row * S_ + t) * H_ + col] = (__bf16)hnew;  // layer output
      } else {  // MODE_Y
        yout[(size_t)row * O_ + col] = v + bias[col];
      }
    }
  }
}

// ---- small helper kernels -------------------------------------------------
__global__ __launch_bounds__(256) void cvt_f32_bf16(const float* __restrict__ in,
                                                    __bf16* __restrict__ out,
                                                    size_t n) {
  size_t i = (size_t)blockIdx.x * blockDim.x + threadIdx.x;
  if (i < n) out[i] = (__bf16)in[i];
}

__global__ __launch_bounds__(256) void init_h_kernel(const float* __restrict__ h0,
                                                     float* __restrict__ hf,
                                                     __bf16* __restrict__ hb,
                                                     int l) {
  int i = blockIdx.x * blockDim.x + threadIdx.x;  // B*H threads
  int b = i / H_, n = i % H_;
  float v = h0[((size_t)b * L_ + l) * H_ + n];
  hf[i] = v;
  hb[i] = (__bf16)v;
}

__global__ __launch_bounds__(256) void store_h_kernel(const float* __restrict__ hf,
                                                      float* __restrict__ out,
                                                      int l) {
  int i = blockIdx.x * blockDim.x + threadIdx.x;  // B*H threads
  int b = i / H_, n = i % H_;
  out[(size_t)B_ * S_ * O_ + ((size_t)b * L_ + l) * H_ + n] = hf[i];
}

// ---------------------------------------------------------------------------
extern "C" void kernel_launch(void* const* d_in, const int* in_sizes, int n_in,
                              void* d_out, int out_size, void* d_ws, size_t ws_size,
                              hipStream_t stream) {
  const float* x  = (const float*)d_in[0];  // [B,S,I]
  const float* h0 = (const float*)d_in[1];  // [B,L,H]
  const float* Wx = (const float*)d_in[2];  // [L,3H,I]
  const float* Wh = (const float*)d_in[3];  // [L,3H,H]
  const float* bh = (const float*)d_in[4];  // [L,3H]
  const float* Wy = (const float*)d_in[5];  // [O,H]
  const float* by = (const float*)d_in[6];  // [O]
  float* out = (float*)d_out;               // [B,S,O] ++ [B,L,H]

  // Workspace carve-up (~251 MB; XP fp32 buffer dominates).
  char* ws = (char*)d_ws;
  size_t off = 0;
  auto carve = [&](size_t bytes) {
    void* p = ws + off;
    off = (off + bytes + 255) & ~(size_t)255;
    return p;
  };
  __bf16* wxb = (__bf16*)carve((size_t)L_ * H3 * I_ * 2);   // 12.6 MB
  __bf16* whb = (__bf16*)carve((size_t)L_ * H3 * H_ * 2);   // 12.6 MB
  __bf16* wyb = (__bf16*)carve((size_t)O_ * H_ * 2);        //  2.1 MB
  __bf16* inb = (__bf16*)carve((size_t)B_ * S_ * I_ * 2);   // 33.6 MB (layer I/O seq)
  float*  xpb = (float*) carve((size_t)B_ * S_ * H3 * 4);   // 201 MB
  float*  hf  = (float*) carve((size_t)B_ * H_ * 4);
  __bf16* hb  = (__bf16*)carve((size_t)B_ * H_ * 2);
  float*  zb  = (float*) carve((size_t)B_ * H_ * 4);
  __bf16* rhb = (__bf16*)carve((size_t)B_ * H_ * 2);

  // 1) fp32 -> bf16 conversions (every call; deterministic)
  {
    size_t n;
    n = (size_t)B_ * S_ * I_;
    cvt_f32_bf16<<<(n + 255) / 256, 256, 0, stream>>>(x, inb, n);
    n = (size_t)L_ * H3 * I_;
    cvt_f32_bf16<<<(n + 255) / 256, 256, 0, stream>>>(Wx, wxb, n);
    n = (size_t)L_ * H3 * H_;
    cvt_f32_bf16<<<(n + 255) / 256, 256, 0, stream>>>(Wh, whb, n);
    n = (size_t)O_ * H_;
    cvt_f32_bf16<<<(n + 255) / 256, 256, 0, stream>>>(Wy, wyb, n);
  }

  const int M = B_ * S_;  // 16384 batched rows (b*S + t)
  for (int l = 0; l < L_; ++l) {
    // 2a) batched input projections: XP = seq_in @ Wx[l]^T  [M, 3H]
    //     LDS-staged A (async), 16x64 wave tiles.
    gru_wmma_gemm<MODE_XP, 4><<<dim3(H3 / 256, M / 32), 256, 0, stream>>>(
        inb, wxb + (size_t)l * H3 * I_,
        nullptr, nullptr, xpb, nullptr, nullptr, nullptr, nullptr, nullptr,
        nullptr, 0);

    // 2b) init recurrent state from h0
    init_h_kernel<<<(B_ * H_) / 256, 256, 0, stream>>>(h0, hf, hb, l);

    const __bf16* whl = whb + (size_t)l * H3 * H_;
    const float*  bhl = bh + (size_t)l * H3;

    // 2c) sequential recurrence (Wh stays L2-resident: 6 MB bf16 / layer).
    for (int t = 0; t < S_; ++t) {
      // z,r gates: [32,2H] = h @ Wh[:2H]^T ; sigmoid; emit z and r*h
      gru_wmma_gemm<MODE_ZR, 1><<<dim3((2 * H_) / 64, 1), 256, 0, stream>>>(
          hb, whl,
          bhl, xpb, nullptr, zb, hf, nullptr, rhb, nullptr, nullptr, t);
      // g gate + state update: [32,H] = (r*h) @ Wh[2H:]^T ; tanh; h_new
      gru_wmma_gemm<MODE_G, 1><<<dim3(H_ / 64, 1), 256, 0, stream>>>(
          rhb, whl + (size_t)2 * H_ * H_,
          bhl, xpb, nullptr, zb, hf, hb, nullptr, inb, nullptr, t);
    }

    // 2d) final hidden state for this layer -> d_out tail
    store_h_kernel<<<(B_ * H_) / 256, 256, 0, stream>>>(hf, out, l);
  }

  // 3) output projection: Y = seq(out of layer L-1) @ Wy^T + by  [M, O] fp32
  gru_wmma_gemm<MODE_Y, 4><<<dim3(O_ / 256, M / 32), 256, 0, stream>>>(
      inb, wyb,
      by, nullptr, nullptr, nullptr, nullptr, nullptr, nullptr, nullptr,
      out, 0);
}
